// NeuralNetworkModel_87308095193599
// MI455X (gfx1250) — matrix operand order repaired and checked by reading, not compiled, not run on hardware
//
#include <hip/hip_runtime.h>
#include <math.h>
#include <stdint.h>

// ---------------------------------------------------------------------------
// MLP: out = gelu(gelu(x@W1^T+b1)@W2^T+b2)@W3^T+b3
// B=16384, D_IN=512, H1=H2=2048, D_OUT=1024
// bf16 WMMA GEMMs (f32 accumulate), async global->LDS staging (no VGPR
// round-trip, no spills), fused bias + exact-erf GELU epilogue, bf16
// intermediates in d_ws.
// ---------------------------------------------------------------------------

#define B_DIM 16384
#define D_IN  512
#define H1D   2048
#define H2D   2048
#define D_OUT 1024

typedef __attribute__((ext_vector_type(16))) __bf16 v16bf;
typedef __attribute__((ext_vector_type(8)))  __bf16 v8bf;
typedef __attribute__((ext_vector_type(4)))  __bf16 v4bf;
typedef __attribute__((ext_vector_type(8)))  float  v8f;

// ---------------------------------------------------------------------------
// f32 -> bf16 conversion (vectorized: float4 in, 4x bf16 out)
// ---------------------------------------------------------------------------
__global__ __launch_bounds__(256) void cvt_f32_to_bf16(
    const float* __restrict__ s, __bf16* __restrict__ d, int n4) {
  int i = blockIdx.x * blockDim.x + threadIdx.x;
  if (i < n4) {
    float4 f = reinterpret_cast<const float4*>(s)[i];
    v4bf o;
    o[0] = (__bf16)f.x; o[1] = (__bf16)f.y;
    o[2] = (__bf16)f.z; o[3] = (__bf16)f.w;
    *reinterpret_cast<v4bf*>(d + (size_t)i * 4) = o;
  }
}

__device__ __forceinline__ float gelu_erf(float v) {
  return 0.5f * v * (1.0f + erff(v * 0.70710678118654752440f));
}

// Truncate a __shared__ pointer to its 32-bit LDS byte offset (aperture rule:
// the LDS address is the low 32 bits of the flat address).
template <typename T>
__device__ __forceinline__ unsigned lds_addr(const T* p) {
  return (unsigned)(uintptr_t)p;
}

// Async copy 16B/lane from global (SGPR base + 32-bit VGPR byte offset, GVS
// mode) directly into LDS (per-lane LDS byte address). Tracked by ASYNCcnt.
__device__ __forceinline__ void async_copy_b128(unsigned lds_byte_addr,
                                                unsigned g_byte_off,
                                                const void* g_base) {
  asm volatile("global_load_async_to_lds_b128 %0, %1, %2"
               :
               : "v"(lds_byte_addr), "v"(g_byte_off), "s"(g_base)
               : "memory");
}

__device__ __forceinline__ void wait_async_then_barrier() {
  asm volatile("s_wait_asynccnt 0x0" ::: "memory");
  __syncthreads();
}

// ---------------------------------------------------------------------------
// Tiled GEMM: C[M,N] = act(A[M,K] @ W[N,K]^T + bias)
// Block: 256 threads (8 waves), tile 128x128, BK=32, double-buffered LDS
// filled by global_load_async_to_lds_b128. Wave grid 4x2: each wave computes
// 32 rows x 64 cols = 2x4 WMMA fragments.
// ---------------------------------------------------------------------------
template <bool APPLY_GELU, typename OUT_T>
__global__ __launch_bounds__(256) void mlp_gemm_wmma_bf16(
    const __bf16* __restrict__ A,    // [M,K] row-major
    const __bf16* __restrict__ W,    // [N,K] row-major
    const float*  __restrict__ bias, // [N]
    OUT_T* __restrict__ C,           // [M,N] row-major
    int M, int N, int K) {
  // LDS row stride 40 bf16 = 80 B: rows 16B-aligned, conflict-free fragment
  // reads (16 distinct rows at the same intra-row offset).
  constexpr int LSTR       = 40;
  constexpr int ROW_BYTES  = LSTR * 2;        // 80
  constexpr int BUF_BYTES  = 128 * ROW_BYTES; // 10240
  __shared__ __attribute__((aligned(16))) __bf16 lA[2][128][LSTR];
  __shared__ __attribute__((aligned(16))) __bf16 lB[2][128][LSTR];

  const int t    = threadIdx.x;
  const int lane = t & 31;
  const int wave = t >> 5;
  const int wm   = wave & 3;   // 0..3 : 32-row slice of the 128-row tile
  const int wn   = wave >> 2;  // 0..1 : 64-col slice of the 128-col tile
  const int half = lane >> 4;  // wave half (ISA WMMA lane layout)
  const int l16  = lane & 15;

  const int rowBase = blockIdx.y * 128;
  const int colBase = blockIdx.x * 128;

  // Staging: thread t covers row lr (+64), 16B chunk lc of the 32-wide K slab.
  const int lr = t >> 2;  // 0..63
  const int lc = t & 3;   // 0..3

  const int KT = K >> 5;  // number of 32-wide K steps

  // Per-thread base addresses for the async copies.
  const unsigned ldsA = lds_addr(&lA[0][0][0]) + lr * ROW_BYTES + lc * 16;
  const unsigned ldsB = lds_addr(&lB[0][0][0]) + lr * ROW_BYTES + lc * 16;
  const unsigned gA   = (unsigned)(((size_t)(rowBase + lr) * K + lc * 8) * 2);
  const unsigned gB   = (unsigned)(((size_t)(colBase + lr) * K + lc * 8) * 2);
  const unsigned gRowStep = (unsigned)(64u * K * 2);  // 64 rows down

  auto issue_tile = [&](int buf, int kt) {
    const unsigned lbufA = ldsA + buf * BUF_BYTES;
    const unsigned lbufB = ldsB + buf * BUF_BYTES;
    const unsigned gk    = (unsigned)kt * 64u;  // kt*32 elements * 2B
#pragma unroll
    for (int p = 0; p < 2; ++p) {
      async_copy_b128(lbufA + p * 64 * ROW_BYTES, gA + gk + p * gRowStep, A);
      async_copy_b128(lbufB + p * 64 * ROW_BYTES, gB + gk + p * gRowStep, W);
    }
  };

  v8f acc[2][4] = {};  // f32 accumulators, 8 VGPRs each

  issue_tile(0, 0);

  for (int kt = 0; kt < KT; ++kt) {
    const int cur = kt & 1;
    wait_async_then_barrier();              // tile `cur` fully in LDS
    if (kt + 1 < KT) issue_tile(cur ^ 1, kt + 1);  // overlap with compute

    // ---- A fragments: 16x32 bf16 (ISA 7.12.2):
    //   lanes 0-15: K {0..7,16..23}; lanes 16-31: K {8..15,24..31}.
    v16bf af[2];
#pragma unroll
    for (int mt = 0; mt < 2; ++mt) {
      const __bf16* pa = &lA[cur][wm * 32 + mt * 16 + l16][0];
      v8bf lo = *reinterpret_cast<const v8bf*>(pa + half * 8);
      v8bf hi = *reinterpret_cast<const v8bf*>(pa + 16 + half * 8);
      af[mt]  = __builtin_shufflevector(lo, hi, 0, 1, 2, 3, 4, 5, 6, 7, 8, 9,
                                        10, 11, 12, 13, 14, 15);
    }
    // ---- B fragments: 32x16 bf16; lanes 0-15: K {0..7,16..23}; lanes 16-31: K {8..15,24..31}.
    v16bf bfr[4];
#pragma unroll
    for (int nt = 0; nt < 4; ++nt) {
      const __bf16* pb = &lB[cur][wn * 64 + nt * 16 + l16][0];
      v8bf lo = *reinterpret_cast<const v8bf*>(pb + half * 8);
      v8bf hi = *reinterpret_cast<const v8bf*>(pb + 16 + half * 8);
      bfr[nt] = __builtin_shufflevector(lo, hi, 0, 1, 2, 3, 4, 5, 6, 7, 8, 9,
                                        10, 11, 12, 13, 14, 15);
    }

#pragma unroll
    for (int mt = 0; mt < 2; ++mt) {
#pragma unroll
      for (int nt = 0; nt < 4; ++nt) {
        acc[mt][nt] = __builtin_amdgcn_wmma_f32_16x16x32_bf16(
            /*neg_a=*/false, af[mt], /*neg_b=*/false, bfr[nt],
            /*c_mod=*/(short)0, acc[mt][nt],
            /*reuse_a=*/false, /*reuse_b=*/false);
      }
    }
  }

  // ---- Epilogue: bias + (optional) exact-erf GELU, direct store.
  // C/D layout: VGPR r -> M = r + 8*half; lane&15 -> N.
#pragma unroll
  for (int mt = 0; mt < 2; ++mt) {
#pragma unroll
    for (int nt = 0; nt < 4; ++nt) {
      const int   col = colBase + wn * 64 + nt * 16 + l16;
      const float bv  = bias[col];
#pragma unroll
      for (int r = 0; r < 8; ++r) {
        const int row = rowBase + wm * 32 + mt * 16 + half * 8 + r;
        float v = acc[mt][nt][r] + bv;
        if (APPLY_GELU) v = gelu_erf(v);
        C[(size_t)row * N + col] = (OUT_T)v;
      }
    }
  }
}

// ---------------------------------------------------------------------------
// Launcher
// ---------------------------------------------------------------------------
extern "C" void kernel_launch(void* const* d_in, const int* in_sizes, int n_in,
                              void* d_out, int out_size, void* d_ws,
                              size_t ws_size, hipStream_t stream) {
  (void)in_sizes; (void)n_in; (void)out_size; (void)ws_size;

  const float* x  = (const float*)d_in[0];
  const float* W1 = (const float*)d_in[1];
  const float* b1 = (const float*)d_in[2];
  const float* W2 = (const float*)d_in[3];
  const float* b2 = (const float*)d_in[4];
  const float* W3 = (const float*)d_in[5];
  const float* b3 = (const float*)d_in[6];
  float* out = (float*)d_out;

  // Workspace carve-up (bf16 copies + bf16 intermediates), 256B aligned.
  char*  ws = (char*)d_ws;
  size_t o  = 0;
  auto carve = [&](size_t elems) -> __bf16* {
    __bf16* p = (__bf16*)(ws + o);
    o = (o + elems * sizeof(__bf16) + 255) & ~(size_t)255;
    return p;
  };
  __bf16* xb  = carve((size_t)B_DIM * D_IN);
  __bf16* w1b = carve((size_t)H1D * D_IN);
  __bf16* w2b = carve((size_t)H2D * H1D);
  __bf16* w3b = carve((size_t)D_OUT * H2D);
  __bf16* h1  = carve((size_t)B_DIM * H1D);
  __bf16* h2  = carve((size_t)B_DIM * H2D);

  auto cvt = [&](const float* s, __bf16* d, size_t n) {
    const int n4 = (int)(n / 4);
    cvt_f32_to_bf16<<<(n4 + 255) / 256, 256, 0, stream>>>(s, d, n4);
  };
  cvt(x,  xb,  (size_t)B_DIM * D_IN);
  cvt(W1, w1b, (size_t)H1D * D_IN);
  cvt(W2, w2b, (size_t)H2D * H1D);
  cvt(W3, w3b, (size_t)D_OUT * H2D);

  // Layer 1: [16384,512] x [2048,512]^T -> gelu -> bf16 h1
  mlp_gemm_wmma_bf16<true, __bf16>
      <<<dim3(H1D / 128, B_DIM / 128), 256, 0, stream>>>(
          xb, w1b, b1, h1, B_DIM, H1D, D_IN);
  // Layer 2: [16384,2048] x [2048,2048]^T -> gelu -> bf16 h2
  mlp_gemm_wmma_bf16<true, __bf16>
      <<<dim3(H2D / 128, B_DIM / 128), 256, 0, stream>>>(
          h1, w2b, b2, h2, B_DIM, H2D, H1D);
  // Layer 3: [16384,2048] x [1024,2048]^T -> f32 out
  mlp_gemm_wmma_bf16<false, float>
      <<<dim3(D_OUT / 128, B_DIM / 128), 256, 0, stream>>>(
          h2, w3b, b3, out, B_DIM, D_OUT, H2D);
}
